// SparseGAT_89240830476322
// MI455X (gfx1250) — matrix operand order repaired and checked by reading, not compiled, not run on hardware
//
#include <hip/hip_runtime.h>
#include <hip/hip_bf16.h>
#include <cstdint>

// ---------------------------------------------------------------- types
typedef __attribute__((ext_vector_type(16))) __bf16 v16bf;
typedef __attribute__((ext_vector_type(8)))  __bf16 v8bf;
typedef __attribute__((ext_vector_type(8)))  float  v8f;

#define GAT_N      50000
#define GAT_E      640000
#define GAT_H      4
#define NEG_SLOPE  0.2f

// ---------------------------------------------------------------- helpers
__device__ __forceinline__ __bf16 f2bf(float f) {
    unsigned u = __float_as_uint(f);
    u = u + 0x7FFFu + ((u >> 16) & 1u);            // round-to-nearest-even
    unsigned short s = (unsigned short)(u >> 16);
    __bf16 r;
    __builtin_memcpy(&r, &s, 2);
    return r;
}

__device__ __forceinline__ void atomicMaxF(float* addr, float v) {
    // ordered-int trick: works with -inf init for mixed-sign values
    if (v >= 0.f) atomicMax((int*)addr, __float_as_int(v));
    else          atomicMin((unsigned int*)addr, (unsigned int)__float_as_int(v));
}

// ---------------------------------------------------------------- utility kernels
__global__ void fill_f32(float* __restrict__ p, float v, size_t n) {
    size_t t = (size_t)blockIdx.x * blockDim.x + threadIdx.x;
    if (t < n) p[t] = v;
}

__global__ void cvt_bf16(const float* __restrict__ in, __bf16* __restrict__ out, size_t n) {
    size_t t = (size_t)blockIdx.x * blockDim.x + threadIdx.x;
    if (t < n) out[t] = f2bf(in[t]);
}

// Repack W [K, Nout] row-major f32 -> WMMA B-fragment layout, bf16.
// Bp[tile*512 + lane*16 + e] = W[(kt*32 + (lane<16?0:8) + (e<8?e:8+e)), tn*16 + (lane&15)]
__global__ void repack_w(const float* __restrict__ W, __bf16* __restrict__ Bp, int K, int Nout) {
    int tid = blockIdx.x * blockDim.x + threadIdx.x;
    if (tid >= K * Nout) return;
    int ntN  = Nout >> 4;
    int e    = tid & 15;
    int lane = (tid >> 4) & 31;
    int tile = tid >> 9;
    int tn   = tile % ntN;
    int kt   = tile / ntN;
    int k = kt * 32 + ((lane < 16) ? 0 : 8) + ((e < 8) ? e : 8 + e);
    int n = tn * 16 + (lane & 15);
    Bp[tid] = f2bf(W[(size_t)k * Nout + n]);
}

// ---------------------------------------------------------------- WMMA GEMM
// C[M,Nout] = A[M,K] @ B[K,Nout], bf16 inputs, f32 accumulate.
// One wave per 16x16 output tile. M % 16 == 0, K % 32 == 0, Nout % 16 == 0.
__global__ __launch_bounds__(256)
void wmma_gemm_bf16(const __bf16* __restrict__ A, const __bf16* __restrict__ Bp,
                    float* __restrict__ C, int M, int K, int Nout) {
    const int lane = threadIdx.x & 31;
    const int wave = (blockIdx.x * blockDim.x + threadIdx.x) >> 5;
    const int ntN  = Nout >> 4;
    const int tm   = wave / ntN;
    const int tn   = wave - tm * ntN;
    if (tm * 16 >= M) return;                     // wave-uniform: EXEC stays all-ones

    const int mrow  = tm * 16 + (lane & 15);
    const int aoff  = (lane < 16) ? 0 : 8;        // 16-bit A layout: lanes 16-31 take K+8
    const __bf16* arow  = A  + (size_t)mrow * K + aoff;
    const __bf16* btile = Bp + ((size_t)tn * 32 + lane) * 16;
    const size_t  bstride = (size_t)ntN * 512;    // per-K-tile stride in packed B

    v8f acc = {};
    const int kt_count = K >> 5;
    for (int kt = 0; kt < kt_count; ++kt) {
        union { v16bf v; v8bf h[2]; } af, bf;
        af.h[0] = *(const v8bf*)(arow);           // K = aoff .. aoff+7
        af.h[1] = *(const v8bf*)(arow + 16);      // K = aoff+16 .. aoff+23
        bf.h[0] = *(const v8bf*)(btile);
        bf.h[1] = *(const v8bf*)(btile + 8);
        __builtin_prefetch(arow + 32, 0, 0);      // global_prefetch_b8 next K tile
        acc = __builtin_amdgcn_wmma_f32_16x16x32_bf16(
                  false, af.v, false, bf.v, (short)0, acc, false, false);
        arow  += 32;
        btile += bstride;
    }

    const int cn    = tn * 16 + (lane & 15);
    const int rbase = tm * 16 + ((lane < 16) ? 0 : 8);
#pragma unroll
    for (int r = 0; r < 8; ++r)
        C[(size_t)(rbase + r) * Nout + cn] = acc[r];
}

// ---------------------------------------------------------------- attention
// s_src[n,h] = sum_c h[n,h,c]*a_src[h,c] ; same for dst
__global__ void attn_scores(const float* __restrict__ h, const float* __restrict__ a_src,
                            const float* __restrict__ a_dst, float* __restrict__ ssrc,
                            float* __restrict__ sdst, int n, int C) {
    int t = blockIdx.x * blockDim.x + threadIdx.x;          // node*4 + head
    if (t >= n * GAT_H) return;
    int hd   = t & 3;
    int node = t >> 2;
    const float* hv = h + (size_t)node * (GAT_H * C) + hd * C;
    const float* as = a_src + hd * C;
    const float* ad = a_dst + hd * C;
    float s1 = 0.f, s2 = 0.f;
    for (int c = 0; c < C; ++c) { s1 += hv[c] * as[c]; s2 += hv[c] * ad[c]; }
    ssrc[t] = s1;
    sdst[t] = s2;
}

// pass 1: e = lrelu(s_src[src]+s_dst[dst]); store e; segment-max into emax[dst]
__global__ void edge_logits(const long long* __restrict__ ei, const float* __restrict__ ssrc,
                            const float* __restrict__ sdst, float* __restrict__ ebuf,
                            float* __restrict__ emax, int E, int n) {
    int i = blockIdx.x * blockDim.x + threadIdx.x;
    if (i >= E + n) return;
    int s = (i < E) ? (int)ei[i]     : (i - E);
    int d = (i < E) ? (int)ei[E + i] : (i - E);
#pragma unroll
    for (int hd = 0; hd < GAT_H; ++hd) {
        float e = ssrc[s * GAT_H + hd] + sdst[d * GAT_H + hd];
        e = (e > 0.f) ? e : NEG_SLOPE * e;
        ebuf[(size_t)i * GAT_H + hd] = e;
        atomicMaxF(&emax[d * GAT_H + hd], e);
    }
}

// pass 2: ex = exp(e - emax[dst]); store ex; segment-sum into denom[dst]
__global__ void edge_exp_k(const long long* __restrict__ ei, float* __restrict__ ebuf,
                           const float* __restrict__ emax, float* __restrict__ denom,
                           int E, int n) {
    int i = blockIdx.x * blockDim.x + threadIdx.x;
    if (i >= E + n) return;
    int d = (i < E) ? (int)ei[E + i] : (i - E);
#pragma unroll
    for (int hd = 0; hd < GAT_H; ++hd) {
        float ex = __expf(ebuf[(size_t)i * GAT_H + hd] - emax[d * GAT_H + hd]);
        ebuf[(size_t)i * GAT_H + hd] = ex;
        atomicAdd(&denom[d * GAT_H + hd], ex);
    }
}

// pass 3: alpha = ex / denom[dst]
__global__ void edge_alpha_k(const long long* __restrict__ ei, float* __restrict__ ebuf,
                             const float* __restrict__ denom, int E, int n) {
    int i = blockIdx.x * blockDim.x + threadIdx.x;
    if (i >= E + n) return;
    int d = (i < E) ? (int)ei[E + i] : (i - E);
#pragma unroll
    for (int hd = 0; hd < GAT_H; ++hd)
        ebuf[(size_t)i * GAT_H + hd] /= denom[d * GAT_H + hd];
}

// pass 4: out[dst, j] += h[src, j] * alpha[edge, head(j)]   (j = h*C + c)
__global__ void edge_scatter(const long long* __restrict__ ei, const float* __restrict__ alpha,
                             const float* __restrict__ hfeat, float* __restrict__ out,
                             int E, int n, int HC, int C) {
    size_t t = (size_t)blockIdx.x * blockDim.x + threadIdx.x;
    size_t total = (size_t)(E + n) * HC;
    if (t >= total) return;
    int i  = (int)(t / HC);
    int j  = (int)(t - (size_t)i * HC);
    int hd = j / C;
    int s = (i < E) ? (int)ei[i]     : (i - E);
    int d = (i < E) ? (int)ei[E + i] : (i - E);
    float a = alpha[(size_t)i * GAT_H + hd];
    atomicAdd(&out[(size_t)d * HC + j], hfeat[(size_t)s * HC + j] * a);
}

// out = leaky_relu(out + b)
__global__ void bias_act(float* __restrict__ h, const float* __restrict__ b, int n, int HC) {
    size_t t = (size_t)blockIdx.x * blockDim.x + threadIdx.x;
    if (t >= (size_t)n * HC) return;
    int j = (int)(t % HC);
    float v = h[t] + b[j];
    h[t] = (v > 0.f) ? v : NEG_SLOPE * v;
}

// global mean pool: accumulate sums & counts per graph
__global__ void pool_kernel(const float* __restrict__ h, const long long* __restrict__ batch,
                            float* __restrict__ sums, float* __restrict__ cnts, int n, int HC) {
    int node = blockIdx.x * blockDim.x + threadIdx.x;
    if (node >= n) return;
    int g = (int)batch[node];
    for (int j = 0; j < HC; ++j)
        atomicAdd(&sums[g * HC + j], h[(size_t)node * HC + j]);
    atomicAdd(&cnts[g], 1.0f);
}

// out[g,o] = sum_k (sums[g,k]/max(cnt,1)) * Wfc[k,o] + bfc[o]
__global__ void fc_kernel(const float* __restrict__ sums, const float* __restrict__ cnts,
                          const float* __restrict__ Wfc, const float* __restrict__ bfc,
                          float* __restrict__ out, int HC, int OD, int G) {
    int t = blockIdx.x * blockDim.x + threadIdx.x;
    if (t >= G * OD) return;
    int g = t / OD, o = t - g * OD;
    float cnt = cnts[g];
    cnt = (cnt > 1.f) ? cnt : 1.f;
    float acc = bfc[o];
    for (int k = 0; k < HC; ++k)
        acc += (sums[g * HC + k] / cnt) * Wfc[k * OD + o];
    out[t] = acc;
}

// ---------------------------------------------------------------- host
extern "C" void kernel_launch(void* const* d_in, const int* in_sizes, int n_in,
                              void* d_out, int out_size, void* d_ws, size_t ws_size,
                              hipStream_t stream) {
    const int N = GAT_N, E = GAT_E;
    const int Ep = E + N;

    const float*     x     = (const float*)d_in[0];
    const long long* ei    = (const long long*)d_in[1];
    const long long* batch = (const long long*)d_in[2];
    const float* Wl[4]  = {(const float*)d_in[3],  (const float*)d_in[7],
                           (const float*)d_in[11], (const float*)d_in[15]};
    const float* Asl[4] = {(const float*)d_in[4],  (const float*)d_in[8],
                           (const float*)d_in[12], (const float*)d_in[16]};
    const float* Adl[4] = {(const float*)d_in[5],  (const float*)d_in[9],
                           (const float*)d_in[13], (const float*)d_in[17]};
    const float* Bl[4]  = {(const float*)d_in[6],  (const float*)d_in[10],
                           (const float*)d_in[14], (const float*)d_in[18]};
    const float* Wfc = (const float*)d_in[19];
    const float* bfc = (const float*)d_in[20];

    const int Kdims[4] = {128, 256, 128, 64};   // input dim per layer
    const int Cdims[4] = {64, 32, 16, 8};       // per-head out channels

    // ---- carve workspace
    char* p = (char*)d_ws;
    auto carve = [&](size_t bytes) -> char* {
        char* r = p;
        p += (bytes + 255) & ~(size_t)255;
        return r;
    };
    float*  bufA  = (float*)carve((size_t)N * 256 * 4);   // GEMM output h
    float*  bufB  = (float*)carve((size_t)N * 256 * 4);   // scatter output / next input
    __bf16* xbf   = (__bf16*)carve((size_t)N * 256 * 2);  // bf16 activations
    __bf16* wp    = (__bf16*)carve((size_t)64 * 1024);    // packed weights (<=64KB)
    float*  ssrc  = (float*)carve((size_t)N * GAT_H * 4);
    float*  sdst  = (float*)carve((size_t)N * GAT_H * 4);
    float*  emax  = (float*)carve((size_t)N * GAT_H * 4);
    float*  denom = (float*)carve((size_t)N * GAT_H * 4);
    float*  ebuf  = (float*)carve((size_t)Ep * GAT_H * 4);
    float*  sums  = (float*)carve((size_t)64 * 32 * 4);
    float*  cnts  = (float*)carve((size_t)64 * 4);

    auto blocks = [](size_t n) { return (unsigned)((n + 255) / 256); };

    const float* xin = x;
    for (int l = 0; l < 4; ++l) {
        const int K = Kdims[l], C = Cdims[l], HC = GAT_H * C;

        // activations -> bf16 ; weights -> packed bf16 B-fragments
        size_t nconv = (size_t)N * K;
        cvt_bf16<<<blocks(nconv), 256, 0, stream>>>(xin, xbf, nconv);
        repack_w<<<blocks((size_t)K * HC), 256, 0, stream>>>(Wl[l], wp, K, HC);

        // h = x @ W  (bf16 WMMA, f32 accumulate)
        int waves = (N / 16) * (HC / 16);
        wmma_gemm_bf16<<<blocks((size_t)waves * 32), 256, 0, stream>>>(xbf, wp, bufA, N, K, HC);

        // attention scores + segment softmax over edges
        attn_scores<<<blocks((size_t)N * GAT_H), 256, 0, stream>>>(bufA, Asl[l], Adl[l],
                                                                   ssrc, sdst, N, C);
        fill_f32<<<blocks((size_t)N * GAT_H), 256, 0, stream>>>(emax, -__builtin_inff(),
                                                                (size_t)N * GAT_H);
        fill_f32<<<blocks((size_t)N * GAT_H), 256, 0, stream>>>(denom, 0.f, (size_t)N * GAT_H);
        edge_logits<<<blocks((size_t)Ep), 256, 0, stream>>>(ei, ssrc, sdst, ebuf, emax, E, N);
        edge_exp_k<<<blocks((size_t)Ep), 256, 0, stream>>>(ei, ebuf, emax, denom, E, N);
        edge_alpha_k<<<blocks((size_t)Ep), 256, 0, stream>>>(ei, ebuf, denom, E, N);

        // weighted scatter-add into bufB, then bias + leaky-relu
        fill_f32<<<blocks((size_t)N * HC), 256, 0, stream>>>(bufB, 0.f, (size_t)N * HC);
        edge_scatter<<<blocks((size_t)Ep * HC), 256, 0, stream>>>(ei, ebuf, bufA, bufB,
                                                                  E, N, HC, C);
        bias_act<<<blocks((size_t)N * HC), 256, 0, stream>>>(bufB, Bl[l], N, HC);

        xin = bufB;   // next layer's input (bufB re-consumed after cvt_bf16)
    }

    // global mean pool + FC head
    fill_f32<<<blocks((size_t)64 * 32), 256, 0, stream>>>(sums, 0.f, (size_t)64 * 32);
    fill_f32<<<1, 64, 0, stream>>>(cnts, 0.f, 64);
    pool_kernel<<<blocks((size_t)N), 256, 0, stream>>>(bufB, batch, sums, cnts, N, 32);
    fc_kernel<<<2, 256, 0, stream>>>(sums, cnts, Wfc, bfc, (float*)d_out, 32, 8, 64);
}
